// GraphVAE_38826504356649
// MI455X (gfx1250) — matrix operand order repaired and checked by reading, not compiled, not run on hardware
//
#include <hip/hip_runtime.h>
#include <hip/hip_bf16.h>
#include <math.h>

typedef __attribute__((ext_vector_type(16))) _Float16 v16h;
typedef __attribute__((ext_vector_type(8)))  float    v8f;
typedef __attribute__((ext_vector_type(2)))  float    v2f;

#define NN 1536
#define HH 32
#define LL 16
#define FF 8
#define BB 2
#define PPAIRS 1178880LL   // N*(N-1)/2
#define PTILES 73680       // PPAIRS / 16 (exact)

__device__ __forceinline__ long long triu_off(long long i) {
  // number of pairs (r,c), r<c, with r < i  ==  i*(2N-1-i)/2
  return i * (2LL * NN - 1 - i) / 2;
}

// ---------------------------------------------------------------------------
// 1) weighted degree (with self-loop fill) -> dinv = deg^-1/2, fix = (diag==0)
// ---------------------------------------------------------------------------
__global__ void k_degree(const float* __restrict__ A, float* __restrict__ dinv,
                         float* __restrict__ fix) {
  int i = blockIdx.x, b = blockIdx.y;
  const float* row = A + ((size_t)b * NN + i) * NN;
  float s = 0.f;
  for (int j = threadIdx.x; j < NN; j += blockDim.x) s += row[j];
  __shared__ float red[256];
  red[threadIdx.x] = s;
  __syncthreads();
  for (int w = 128; w > 0; w >>= 1) {
    if ((int)threadIdx.x < w) red[threadIdx.x] += red[threadIdx.x + w];
    __syncthreads();
  }
  if (threadIdx.x == 0) {
    float aii = row[i];
    float fx  = (aii == 0.f) ? 1.f : 0.f;   // A_hat = A + eye * fx
    float deg = red[0] + fx;                // rowsum(A_hat)
    dinv[b * NN + i] = (deg > 0.f) ? rsqrtf(deg) : 0.f;
    fix [b * NN + i] = fx;
  }
}

// ---------------------------------------------------------------------------
// 2) T = (X @ W) * dinv   (small K: 8 or 32) -- VALU, trivial cost
// ---------------------------------------------------------------------------
__global__ void k_linear_scale(const float* __restrict__ X, const float* __restrict__ W,
                               const float* __restrict__ dinv, float* __restrict__ T,
                               int K, int xBatched) {
  int idx = blockIdx.x * blockDim.x + threadIdx.x;
  if (idx >= BB * NN * HH) return;
  int h = idx & (HH - 1);
  int n = (idx / HH) % NN;
  int b = idx / (HH * NN);
  const float* xr = X + ((size_t)(xBatched ? (b * NN + n) : n)) * K;
  float acc = 0.f;
  for (int k = 0; k < K; ++k) acc += xr[k] * W[k * HH + h];
  T[idx] = acc * dinv[b * NN + n];
}

// ---------------------------------------------------------------------------
// 3) Y = A @ T via V_WMMA_F32_16X16X4_F32; epilogue: diag fix, dinv scale,
//    bias, relu, optional residual.  One wave per 16x16 output tile.
// ---------------------------------------------------------------------------
__global__ void k_prop_wmma(const float* __restrict__ A, const float* __restrict__ T,
                            const float* __restrict__ dinv, const float* __restrict__ fix,
                            const float* __restrict__ bias, const float* __restrict__ skip,
                            float* __restrict__ out) {
  int rtile = blockIdx.x, ctile = blockIdx.y, b = blockIdx.z;
  int lane = threadIdx.x;
  int hi   = lane >> 4;        // 0: low 16 lanes, 1: high 16 lanes
  int nl   = lane & 15;
  int n    = ctile * 16 + nl;  // output column (0..31)
  int ri   = rtile * 16;       // output row base
  const float* Arow = A + ((size_t)b * NN + ri + nl) * NN;  // A-frag row M = nl
  const float* Tb   = T + (size_t)b * NN * HH;

  v8f acc = {};
  for (int k0 = 0; k0 < NN; k0 += 4) {
    int kk = k0 + 2 * hi;
    if ((k0 & 127) == 0) __builtin_prefetch(Arow + kk + 512, 0, 0);
    v2f a, bf;
    a.x  = Arow[kk];
    a.y  = Arow[kk + 1];
    bf.x = Tb[(size_t)kk * HH + n];
    bf.y = Tb[(size_t)(kk + 1) * HH + n];
    acc = __builtin_amdgcn_wmma_f32_16x16x4_f32(false, a, false, bf,
                                                (short)0, acc, false, false);
  }
  for (int r = 0; r < 8; ++r) {
    int i = ri + r + 8 * hi;
    size_t bi = (size_t)b * NN + i;
    float val = acc[r] + fix[bi] * Tb[(size_t)i * HH + n];  // + diag self-loop term
    float hv  = fmaxf(dinv[bi] * val + bias[n], 0.f);
    if (skip) hv += skip[bi * HH + n];
    out[bi * HH + n] = hv;
  }
}

// ---------------------------------------------------------------------------
// 4) mu / logvar heads (K=32 -> 16), VALU
// ---------------------------------------------------------------------------
__global__ void k_mulv(const float* __restrict__ h3, const float* __restrict__ Wmu,
                       const float* __restrict__ bmu, const float* __restrict__ Wlv,
                       const float* __restrict__ blv, float* __restrict__ mu,
                       float* __restrict__ lv) {
  int idx = blockIdx.x * blockDim.x + threadIdx.x;
  if (idx >= BB * NN * LL) return;
  int l  = idx & (LL - 1);
  int bn = idx / LL;
  const float* hr = h3 + (size_t)bn * HH;
  float m = bmu[l], v = blv[l];
  for (int k = 0; k < HH; ++k) {
    m += hr[k] * Wmu[k * LL + l];
    v += hr[k] * Wlv[k * LL + l];
  }
  mu[idx] = m;
  lv[idx] = v;
}

// ---------------------------------------------------------------------------
// 5) zero the diagonal of adj_pred (off-diagonals fully covered by edge kernel)
// ---------------------------------------------------------------------------
__global__ void k_zdiag(float* __restrict__ adj) {
  int idx = blockIdx.x * blockDim.x + threadIdx.x;
  if (idx >= BB * NN) return;
  int b = idx / NN, i = idx % NN;
  adj[(size_t)b * NN * NN + (size_t)i * NN + i] = 0.f;
}

// ---------------------------------------------------------------------------
// 6) edge MLP over all pairs: 16 pairs per wave per iteration.
//    layer1: [16x32]@[32x64] -> 4x v_wmma_f32_16x16x32_f16
//    layer2: [16x64]@[64x32] -> 4x chained wmma (via LDS relayout)
//    layer3: VALU dot(32) + shfl_xor reduce + sigmoid + symmetric scatter
// ---------------------------------------------------------------------------
__global__ void __launch_bounds__(256) k_edge(
    const float* __restrict__ mu,
    const float* __restrict__ E1, const float* __restrict__ e1,
    const float* __restrict__ E2, const float* __restrict__ e2,
    const float* __restrict__ E3, const float* __restrict__ e3,
    float* __restrict__ adj) {
  __shared__ _Float16 sh[8 * 16 * 64];  // one 16x64 f16 tile per wave
  int lane = threadIdx.x & 31;
  int wib  = threadIdx.x >> 5;
  _Float16* myLds = sh + wib * (16 * 64);
  int hi = lane >> 4;
  int nl = lane & 15;

  // ---- hoisted weight fragments (B-layout: N = nl, K split across halves) ----
  v16h bE1[4];                      // E1: 32x64, four 32x16 N-tiles
  for (int t = 0; t < 4; ++t)
    for (int h = 0; h < 16; ++h) {
      int K = h + 16 * hi;
      bE1[t][h] = (_Float16)E1[K * 64 + t * 16 + nl];
    }
  v16h bE2[2][2];                   // E2: 64x32, [N-tile][K-half]
  for (int t = 0; t < 2; ++t)
    for (int q = 0; q < 2; ++q)
      for (int h = 0; h < 16; ++h) {
        int K = h + 16 * hi + 32 * q;
        bE2[t][q][h] = (_Float16)E2[K * 32 + t * 16 + nl];
      }
  float e1v[4]; for (int t = 0; t < 4; ++t) e1v[t] = e1[t * 16 + nl];
  float e2v[2]; for (int t = 0; t < 2; ++t) e2v[t] = e2[t * 16 + nl];
  float E3a = E3[nl], E3b = E3[16 + nl], e3s = e3[0];

  int wglobal = blockIdx.x * (blockDim.x >> 5) + wib;
  int nWaves  = gridDim.x * (blockDim.x >> 5);
  const int TT = BB * PTILES;

  for (int t = wglobal; t < TT; t += nWaves) {
    int b    = t / PTILES;
    int tile = t - b * PTILES;
    long long p = (long long)tile * 16 + nl;   // this lane's pair (row M = nl)

    // invert the triangular index: largest i with triu_off(i) <= p
    double s    = 2.0 * NN - 1.0;
    double disc = s * s - 8.0 * (double)p;
    long long i = (long long)((s - sqrt(disc)) * 0.5);
    if (i < 0) i = 0;
    if (i > NN - 2) i = NN - 2;
    while (i > 0 && triu_off(i) > p) --i;
    while (i < NN - 2 && triu_off(i + 1) <= p) ++i;
    int j = (int)(i + 1 + (p - triu_off(i)));

    // A-fragment: feats = [z_i (16) | z_j (16)] in f16
    const float* zi = mu + ((size_t)b * NN + i) * LL;
    const float* zj = mu + ((size_t)b * NN + j) * LL;
    v16h a;
    for (int h = 0; h < 16; ++h) {
      int K = h + 16 * hi;
      float v = (K < 16) ? zi[K] : zj[K - 16];
      a[h] = (_Float16)v;
    }

    // layer 1: h1 = relu(feats @ E1 + e1)   [16 x 64]
    v8f acc1[4];
    for (int t4 = 0; t4 < 4; ++t4) {
      v8f c;
      for (int r = 0; r < 8; ++r) c[r] = e1v[t4];
      acc1[t4] = __builtin_amdgcn_wmma_f32_16x16x32_f16(
          false, a, false, bE1[t4], (short)0, c, false, false);
    }

    // relayout h1 -> A-fragment via wave-private LDS (f16, 16x64 row-major)
    for (int t4 = 0; t4 < 4; ++t4)
      for (int r = 0; r < 8; ++r) {
        int M = r + 8 * hi, n = t4 * 16 + nl;
        myLds[M * 64 + n] = (_Float16)fmaxf(acc1[t4][r], 0.f);
      }
    __builtin_amdgcn_wave_barrier();   // LDS is in-order per wave; block reordering
    v16h aLo, aHi;
    for (int h = 0; h < 16; ++h) {
      int K = h + 16 * hi;
      aLo[h] = myLds[nl * 64 + K];
      aHi[h] = myLds[nl * 64 + K + 32];
    }
    __builtin_amdgcn_wave_barrier();

    // layer 2: h2 = relu(h1 @ E2 + e2)   [16 x 32], K=64 chained
    v8f acc2[2];
    for (int t2 = 0; t2 < 2; ++t2) {
      v8f c;
      for (int r = 0; r < 8; ++r) c[r] = e2v[t2];
      c = __builtin_amdgcn_wmma_f32_16x16x32_f16(
          false, aLo, false, bE2[t2][0], (short)0, c, false, false);
      c = __builtin_amdgcn_wmma_f32_16x16x32_f16(
          false, aHi, false, bE2[t2][1], (short)0, c, false, false);
      acc2[t2] = c;
    }

    // layer 3: w = sigmoid(relu(h2) . E3 + e3); reduce across 16-lane halves
    float part[8];
    for (int r = 0; r < 8; ++r)
      part[r] = fmaxf(acc2[0][r], 0.f) * E3a + fmaxf(acc2[1][r], 0.f) * E3b;
    for (int m = 1; m <= 8; m <<= 1)
      for (int r = 0; r < 8; ++r)
        part[r] += __shfl_xor(part[r], m, 32);

    // lanes 0..7 own rows 0..7; lanes 24..31 own rows 8..15 (their own (i,j))
    if (lane < 8 || lane >= 24) {
      int r = (lane < 8) ? lane : (lane - 24);
      float w = 1.f / (1.f + __expf(-(part[r] + e3s)));
      float* ab = adj + (size_t)b * NN * NN;
      ab[(size_t)i * NN + j] = w;
      ab[(size_t)j * NN + i] = w;
    }
  }
}

// ---------------------------------------------------------------------------
extern "C" void kernel_launch(void* const* d_in, const int* in_sizes, int n_in,
                              void* d_out, int out_size, void* d_ws, size_t ws_size,
                              hipStream_t stream) {
  const float* A   = (const float*)d_in[0];
  const float* emb = (const float*)d_in[1];
  const float* W1  = (const float*)d_in[2];
  const float* b1  = (const float*)d_in[3];
  const float* W2  = (const float*)d_in[4];
  const float* b2  = (const float*)d_in[5];
  const float* W3  = (const float*)d_in[6];
  const float* b3  = (const float*)d_in[7];
  const float* Wmu = (const float*)d_in[8];
  const float* bmu = (const float*)d_in[9];
  const float* Wlv = (const float*)d_in[10];
  const float* blv = (const float*)d_in[11];
  const float* E1  = (const float*)d_in[12];
  const float* e1  = (const float*)d_in[13];
  const float* E2  = (const float*)d_in[14];
  const float* e2  = (const float*)d_in[15];
  const float* E3  = (const float*)d_in[16];
  const float* e3  = (const float*)d_in[17];

  float* out = (float*)d_out;
  float* adj = out;                              // [B,N,N]
  float* mu  = out + (size_t)BB * NN * NN;       // [B*N, L]
  float* lv  = mu + (size_t)BB * NN * LL;        // [B*N, L]

  float* ws   = (float*)d_ws;
  float* dinv = ws;                  // B*N
  float* fix  = dinv + BB * NN;      // B*N
  float* T    = fix  + BB * NN;      // B*N*H
  float* h1   = T  + (size_t)BB * NN * HH;
  float* h2   = h1 + (size_t)BB * NN * HH;
  float* h3   = h2 + (size_t)BB * NN * HH;

  k_degree<<<dim3(NN, BB), 256, 0, stream>>>(A, dinv, fix);

  int linThreads = BB * NN * HH;
  k_linear_scale<<<(linThreads + 255) / 256, 256, 0, stream>>>(emb, W1, dinv, T, FF, 0);
  k_prop_wmma<<<dim3(NN / 16, HH / 16, BB), 32, 0, stream>>>(A, T, dinv, fix, b1, nullptr, h1);
  k_linear_scale<<<(linThreads + 255) / 256, 256, 0, stream>>>(h1, W2, dinv, T, HH, 1);
  k_prop_wmma<<<dim3(NN / 16, HH / 16, BB), 32, 0, stream>>>(A, T, dinv, fix, b2, h1, h2);
  k_linear_scale<<<(linThreads + 255) / 256, 256, 0, stream>>>(h2, W3, dinv, T, HH, 1);
  k_prop_wmma<<<dim3(NN / 16, HH / 16, BB), 32, 0, stream>>>(A, T, dinv, fix, b3, h2, h3);

  int mlThreads = BB * NN * LL;
  k_mulv<<<(mlThreads + 255) / 256, 256, 0, stream>>>(h3, Wmu, bmu, Wlv, blv, mu, lv);
  k_zdiag<<<(BB * NN + 255) / 256, 256, 0, stream>>>(adj);
  k_edge<<<2048, 256, 0, stream>>>(mu, E1, e1, E2, e2, E3, e3, adj);
}
